// Wav2Vec2BertAttention_23424751632752
// MI455X (gfx1250) — compile-verified, matrix operand order checked by reading, not run on hardware
//
#include <hip/hip_runtime.h>

// ---------------------------------------------------------------------------
// Wav2Vec2-BERT attention for MI455X (gfx1250, wave32, WMMA f32_16x16x32_f16)
//   B=4, S=1024, E=1024, H=16, D=64, NUM_POS=129, SCALING = 1/8
// Pipeline:
//   0) f32 -> f16 pre-convert of hidden_states / qkv_w / out_w (one pass,
//      removes per-tile v_cvt from GEMM hot loops, halves L2 operand bytes)
//   1) QKV GEMM (WMMA, 32x64 wave tile) -> q,k [B,H,S,D] f16, V^T [B,H,D,S] f16
//   2) qd[row,p] = q . dist_emb[p]   (tiny GEMM, VALU)
//   3) attention: WMMA scores + rel-bias gather -> softmax -> WMMA P@V
//      (writes full attn_weights f32 once: the mandatory 256 MB stream)
//   4) out projection GEMM (WMMA) -> f32 attn_output
// ---------------------------------------------------------------------------

typedef _Float16 half_t;
typedef __attribute__((ext_vector_type(16))) _Float16 v16h;
typedef __attribute__((ext_vector_type(8)))  _Float16 v8h;
typedef __attribute__((ext_vector_type(4)))  _Float16 v4h;
typedef __attribute__((ext_vector_type(8)))  float    v8f;
typedef __attribute__((ext_vector_type(4)))  float    v4f;

#define WMMA(a, b, c) \
    __builtin_amdgcn_wmma_f32_16x16x32_f16(false, (a), false, (b), (short)0, (c), false, false)

// --- Fragment loader -------------------------------------------------------
// CDNA5 16-bit 16x32 fragment layout (ISA 7.12.2): lane L holds row (L&15);
// halves h=0..15 map to K = (h<8 ? h : h+8) + ((L>>4)<<3).
// Same per-lane addressing works for the A matrix (row = M) and for the B
// matrix (row = N) when the source is stored row-major [row][K].
__device__ __forceinline__ v16h frag_load_f16(const half_t* __restrict__ base, int ldh) {
    const int lane = threadIdx.x & 31;
    const int r    = lane & 15;
    const int ko   = (lane >> 4) << 3;           // 0 or 8
    const half_t* p = base + (size_t)r * ldh + ko;
    const v8h lo = *reinterpret_cast<const v8h*>(p);        // K = ko..ko+7
    const v8h hi = *reinterpret_cast<const v8h*>(p + 16);   // K = ko+16..ko+23
    v16h f;
#pragma unroll
    for (int i = 0; i < 8; ++i) { f[i] = lo[i]; f[i + 8] = hi[i]; }
    return f;
}

// --- Kernel 0: streaming f32 -> f16 convert (vector4) ----------------------
__global__ __launch_bounds__(256) void cvt_f16_kernel(
    const float* __restrict__ in, half_t* __restrict__ out, int n4) {
    const int i = blockIdx.x * 256 + threadIdx.x;
    if (i < n4) {
        const v4f v = reinterpret_cast<const v4f*>(in)[i];
        v4h h;
#pragma unroll
        for (int j = 0; j < 4; ++j) h[j] = (half_t)v[j];
        reinterpret_cast<v4h*>(out)[i] = h;
    }
}

// --- Kernel 1: fused QKV projection ---------------------------------------
// C[m,n] = sum_k X[m,k] * Wqkv[n,k] + b[n];  M=4096, N=3072, K=1024.
// Wave tile 32x64 (8 WMMAs / K-step). Epilogue scatters to q/k [B,H,S,D] f16
// and vT [B,H,D,S] f16.
__global__ __launch_bounds__(128) void qkv_gemm_kernel(
    const half_t* __restrict__ X, const half_t* __restrict__ W,
    const float* __restrict__ bias,
    half_t* __restrict__ qf, half_t* __restrict__ kf, half_t* __restrict__ vT) {

    const int waveId = blockIdx.x * 4 + (threadIdx.x >> 5);
    const int mt = waveId / 48;            // 128 M-tiles of 32
    const int nt = waveId % 48;            // 48 N-tiles of 64
    const int m0 = mt * 32;
    const int n0 = nt * 64;

    v8f acc[2][4] = {};
    for (int k = 0; k < 1024; k += 32) {
        const v16h a0 = frag_load_f16(X + (size_t)m0 * 1024 + k, 1024);
        const v16h a1 = frag_load_f16(X + (size_t)(m0 + 16) * 1024 + k, 1024);
#pragma unroll
        for (int t = 0; t < 4; ++t) {
            const v16h b = frag_load_f16(W + (size_t)(n0 + t * 16) * 1024 + k, 1024);
            acc[0][t] = WMMA(a0, b, acc[0][t]);
            acc[1][t] = WMMA(a1, b, acc[1][t]);
        }
    }

    const int lane = threadIdx.x & 31;
    const int nlo  = lane & 15;
    const int mhi  = (lane >> 4) << 3;
#pragma unroll
    for (int rh = 0; rh < 2; ++rh) {
#pragma unroll
        for (int t = 0; t < 4; ++t) {
#pragma unroll
            for (int j = 0; j < 8; ++j) {
                const int m = m0 + rh * 16 + j + mhi;   // token row (b*1024+s)
                const int n = n0 + t * 16 + nlo;        // feature in [0,3072)
                const float val = acc[rh][t][j] + bias[n];
                const int b = m >> 10, s = m & 1023;
                const int sect = n >> 10, nn = n & 1023;
                const int h = nn >> 6, d = nn & 63;
                const size_t bh = (size_t)(b * 16 + h);
                if (sect == 0)      qf[(bh * 1024 + s) * 64 + d] = (half_t)val;
                else if (sect == 1) kf[(bh * 1024 + s) * 64 + d] = (half_t)val;
                else                vT[(bh * 64 + d) * 1024 + s] = (half_t)val;
            }
        }
    }
}

// --- Kernel 2: qd[row,p] = q_row . dist_emb[p]  (row = bh*1024+l, p<129) ---
#define QDSTR 132
__global__ __launch_bounds__(256) void qd_kernel(
    const half_t* __restrict__ qf, const float* __restrict__ de,
    float* __restrict__ qd) {
    const int row  = blockIdx.x * 8 + (threadIdx.x >> 5);
    const int lane = threadIdx.x & 31;
    const half_t* qr = qf + (size_t)row * 64;
    float qv[64];
#pragma unroll
    for (int i = 0; i < 64; ++i) qv[i] = (float)qr[i];
    for (int p = lane; p < 129; p += 32) {
        const float* dr = de + p * 64;
        float acc = 0.f;
#pragma unroll
        for (int i = 0; i < 64; ++i) acc = fmaf(qv[i], dr[i], acc);
        qd[(size_t)row * QDSTR + p] = acc;
    }
}

// --- Kernel 3: attention (scores + bias -> softmax -> P@V) -----------------
// One block per (bh, 16-row q tile). 4 waves: each wave computes scores for a
// 256-wide column strip, then 4 softmax rows, then one 16-wide d-tile of P@V.
#define SSTR 1028   // f32 score row stride (pads banks)
#define PSTR 1056   // f16 prob row stride (multiple of 8 halves -> 16B aligned)

__global__ __launch_bounds__(128) void attn_kernel(
    const half_t* __restrict__ qf, const half_t* __restrict__ kf,
    const half_t* __restrict__ vT, const float* __restrict__ qd,
    float* __restrict__ attn_w, half_t* __restrict__ ctx) {

    __shared__ __align__(16) float  s_sc[16 * SSTR];   // 65.8 KB
    __shared__ __align__(16) half_t s_p [16 * PSTR];   // 33.8 KB
    __shared__ __align__(16) float  s_qd[16 * QDSTR];  //  8.4 KB

    const int blk  = blockIdx.x;        // 0..4095
    const int bh   = blk >> 6;          // 0..63   (b*16+h)
    const int l0   = (blk & 63) * 16;   // q row base
    const int wave = threadIdx.x >> 5;
    const int lane = threadIdx.x & 31;

    // Preload relative-position dot products for these 16 rows.
    const float* qd_base = qd + ((size_t)bh * 1024 + l0) * QDSTR;
    for (int idx = threadIdx.x; idx < 16 * 129; idx += 128) {
        const int rr = idx / 129, pp = idx - rr * 129;
        s_qd[rr * QDSTR + pp] = qd_base[rr * QDSTR + pp];
    }
    __syncthreads();

    // Q fragments (K = 0..31, 32..63), reused across all column blocks.
    const half_t* qbase = qf + ((size_t)bh * 1024 + l0) * 64;
    const v16h a0 = frag_load_f16(qbase, 64);
    const v16h a1 = frag_load_f16(qbase + 32, 64);

    const int nlo = lane & 15;
    const int mhi = (lane >> 4) << 3;
    const float scale = 0.125f;          // HEAD_DIM^-0.5

    // Phase 1: scores for columns [wave*256, wave*256+256)
    for (int cb = 0; cb < 16; ++cb) {
        const int c0 = wave * 256 + cb * 16;
        const half_t* kb = kf + ((size_t)bh * 1024 + c0) * 64;
        const v16h b0 = frag_load_f16(kb, 64);
        const v16h b1 = frag_load_f16(kb + 32, 64);
        v8f c = {};
        c = WMMA(a0, b0, c);
        c = WMMA(a1, b1, c);
#pragma unroll
        for (int j = 0; j < 8; ++j) {
            const int m = j + mhi;                 // row within tile
            const int r = c0 + nlo;                // global key col
            int dd = r - (l0 + m);
            dd = dd < -64 ? -64 : (dd > 64 ? 64 : dd);
            s_sc[m * SSTR + r] = scale * (c[j] + s_qd[m * QDSTR + dd + 64]);
        }
    }
    __syncthreads();

    // Phase 2: softmax (4 rows per wave) + write f32 weights + f16 probs.
    for (int i = 0; i < 4; ++i) {
        const int m = wave * 4 + i;
        float* row = s_sc + m * SSTR;
        float mx = -1e30f;
        for (int t = lane; t < 1024; t += 32) mx = fmaxf(mx, row[t]);
#pragma unroll
        for (int o = 16; o; o >>= 1) mx = fmaxf(mx, __shfl_xor(mx, o, 32));
        float sum = 0.f;
        for (int t = lane; t < 1024; t += 32) {
            const float e = __expf(row[t] - mx);
            row[t] = e;
            sum += e;
        }
#pragma unroll
        for (int o = 16; o; o >>= 1) sum += __shfl_xor(sum, o, 32);
        const float inv = 1.f / sum;
        float* gout = attn_w + ((size_t)bh * 1024 + (l0 + m)) * 1024;
        for (int t = lane; t < 1024; t += 32) {
            const float p = row[t] * inv;
            gout[t] = p;
            s_p[m * PSTR + t] = (half_t)p;
        }
    }
    __syncthreads();

    // Phase 3: P @ V.  Wave owns d-tile [wave*16, wave*16+16); K = 1024.
    v8f acc = {};
    const half_t* vb = vT + ((size_t)bh * 64 + wave * 16) * 1024;
    for (int k = 0; k < 1024; k += 32) {
        const v16h a = frag_load_f16(s_p + k, PSTR);   // probs (LDS)
        const v16h b = frag_load_f16(vb + k, 1024);    // V^T rows (= d)
        acc = WMMA(a, b, acc);
    }
    const int b = bh >> 4, h = bh & 15;
#pragma unroll
    for (int j = 0; j < 8; ++j) {
        const int l = l0 + j + mhi;
        const int d = wave * 16 + nlo;
        ctx[((size_t)b * 1024 + l) * 1024 + h * 64 + d] = (half_t)acc[j];
    }
}

// --- Kernel 4: output projection ------------------------------------------
// Y[m,n] = sum_k ctx[m,k] * Wout[n,k] + b[n];  M=4096, N=1024, K=1024 -> f32.
// Wave tile 32x64.
__global__ __launch_bounds__(128) void out_gemm_kernel(
    const half_t* __restrict__ ctx, const half_t* __restrict__ W,
    const float* __restrict__ bias, float* __restrict__ Y) {

    const int waveId = blockIdx.x * 4 + (threadIdx.x >> 5);
    const int mt = waveId >> 4;            // 128 M-tiles of 32
    const int nt = waveId & 15;            // 16 N-tiles of 64
    const int m0 = mt * 32;
    const int n0 = nt * 64;

    v8f acc[2][4] = {};
    for (int k = 0; k < 1024; k += 32) {
        const v16h a0 = frag_load_f16(ctx + (size_t)m0 * 1024 + k, 1024);
        const v16h a1 = frag_load_f16(ctx + (size_t)(m0 + 16) * 1024 + k, 1024);
#pragma unroll
        for (int t = 0; t < 4; ++t) {
            const v16h b = frag_load_f16(W + (size_t)(n0 + t * 16) * 1024 + k, 1024);
            acc[0][t] = WMMA(a0, b, acc[0][t]);
            acc[1][t] = WMMA(a1, b, acc[1][t]);
        }
    }
    const int lane = threadIdx.x & 31;
    const int nlo  = lane & 15;
    const int mhi  = (lane >> 4) << 3;
#pragma unroll
    for (int rh = 0; rh < 2; ++rh) {
#pragma unroll
        for (int t = 0; t < 4; ++t) {
#pragma unroll
            for (int j = 0; j < 8; ++j) {
                const int n = n0 + t * 16 + nlo;
                Y[(size_t)(m0 + rh * 16 + j + mhi) * 1024 + n] = acc[rh][t][j] + bias[n];
            }
        }
    }
}

// --- Host launcher ---------------------------------------------------------
extern "C" void kernel_launch(void* const* d_in, const int* in_sizes, int n_in,
                              void* d_out, int out_size, void* d_ws, size_t ws_size,
                              hipStream_t stream) {
    const float* hs     = (const float*)d_in[0];   // [4,1024,1024]
    const float* qkv_w  = (const float*)d_in[1];   // [3072,1024]
    const float* qkv_b  = (const float*)d_in[2];   // [3072]
    const float* out_w  = (const float*)d_in[3];   // [1024,1024]
    const float* out_b  = (const float*)d_in[4];   // [1024]
    const float* dist_e = (const float*)d_in[5];   // [129,64]

    char* ws = (char*)d_ws;
    half_t* qf     = (half_t*)(ws);                        // 8 MB  [B,H,S,D]
    half_t* kf     = (half_t*)(ws + ((size_t)8  << 20));   // 8 MB  [B,H,S,D]
    half_t* vT     = (half_t*)(ws + ((size_t)16 << 20));   // 8 MB  [B,H,D,S]
    half_t* ctx    = (half_t*)(ws + ((size_t)24 << 20));   // 8 MB  [B,S,E]
    float*  qd     = (float*) (ws + ((size_t)32 << 20));   // ~33 MB [bh*S,132]
    half_t* hs16   = (half_t*)(ws + ((size_t)66 << 20));   // 8 MB
    half_t* qkvw16 = (half_t*)(ws + ((size_t)74 << 20));   // 6 MB
    half_t* outw16 = (half_t*)(ws + ((size_t)80 << 20));   // 2 MB

    float* attn_out = (float*)d_out;                          // [B,S,E]
    float* attn_w   = attn_out + (size_t)4 * 1024 * 1024;     // [B,H,S,S]

    // 0) pre-convert f32 operands to f16 (single streaming pass each)
    cvt_f16_kernel<<<(4 * 1024 * 1024 / 4 + 255) / 256, 256, 0, stream>>>(hs,    hs16,   4 * 1024 * 1024 / 4);
    cvt_f16_kernel<<<(3 * 1024 * 1024 / 4 + 255) / 256, 256, 0, stream>>>(qkv_w, qkvw16, 3 * 1024 * 1024 / 4);
    cvt_f16_kernel<<<(1024 * 1024 / 4 + 255) / 256, 256, 0, stream>>>(out_w, outw16, 1024 * 1024 / 4);

    // 1..4) pipeline
    qkv_gemm_kernel<<<1536, 128, 0, stream>>>(hs16, qkvw16, qkv_b, qf, kf, vT);
    qd_kernel      <<<8192, 256, 0, stream>>>(qf, dist_e, qd);
    attn_kernel    <<<4096, 128, 0, stream>>>(qf, kf, vT, qd, attn_w, ctx);
    out_gemm_kernel<<<512, 128, 0, stream>>>(ctx, outw16, out_b, attn_out);
}
